// VDP_first_Conv_59433757442545
// MI455X (gfx1250) — compile-verified, matrix operand order checked by reading, not run on hardware
//
#include <hip/hip_runtime.h>

typedef __attribute__((ext_vector_type(16))) _Float16 v16h;
typedef __attribute__((ext_vector_type(8)))  _Float16 v8h;
typedef __attribute__((ext_vector_type(8)))  float    v8f;

#define NB    128   // batch
#define H_    128
#define W_    128
#define C_    3
#define KS_   5
#define KN_   64    // output channels
#define HO_   124
#define WO_   124
#define KDIM  75    // 5*5*3
#define KPAD  96    // 12 K-blocks of 8; 3 x K=32 WMMA steps

// Fragment-native K-block permutation: the (kc,hi,ehalf) slot at index cs
// holds K-block blk = cs with bits 0 and 1 swapped (involution).
__device__ __forceinline__ int blkslot(int blk) {
    return (blk & ~3) | ((blk & 1) << 1) | ((blk >> 1) & 1);
}

__device__ __forceinline__ float softplus_f(float x) { return log1pf(expf(x)); }

// ---------------------------------------------------------------------------
// KL regularizer (one tiny block).
// ---------------------------------------------------------------------------
__global__ void kl_kernel(const float* __restrict__ w_mu,
                          const float* __restrict__ w_sigma,
                          float* __restrict__ kl_out) {
    __shared__ float terms[KN_];
    const int n = threadIdx.x;          // 64 threads
    float ss = 0.f;
    for (int i = 0; i < KDIM; ++i) { float v = w_mu[i * KN_ + n]; ss += v * v; }
    const float ws = w_sigma[n];
    terms[n] = 1.f + ws - softplus_f(ws) - ss * (1.f / (float)KDIM);
    __syncthreads();
    if (n == 0) {
        float acc = 0.f;
        for (int i = 0; i < KN_; ++i) acc += terms[i];
        float kl = -acc / (float)KN_;
        if (isnan(kl) || isinf(kl)) kl = 1e-5f;
        *kl_out = kl;
    }
}

// ---------------------------------------------------------------------------
// Fused conv (implicit GEMM, v_wmma_f32_16x16x32_f16) + Sigma_out.
// One block per (image, output row); 128 threads = 4 waves.
// LDS holds fragment-native im2col (s_a) and weights (s_wf) so every WMMA
// operand is exactly two ds_load_b128 per lane.
// ---------------------------------------------------------------------------
__global__ void __launch_bounds__(128)
vdp_conv_kernel(const float* __restrict__ in,
                const float* __restrict__ w_mu,
                const float* __restrict__ w_sigma,
                float* __restrict__ mu_out,
                float* __restrict__ sigma_out) {
    __shared__ __align__(16) _Float16 s_a[128 * KPAD];   // im2col tile, 24 KB
    __shared__ __align__(16) _Float16 s_wf[KN_ * KPAD];  // weights, 12 KB
    __shared__ float s_sq[KS_ * W_];                     // per-(row,x) sum_c x^2
    __shared__ float s_ss[WO_];                          // 5x5 window sum-sq
    __shared__ float s_sp[KN_];                          // softplus(w_sigma)

    const int tid = threadIdx.x;
    const int oh  = blockIdx.x;   // 0..123
    const int b   = blockIdx.y;   // 0..127

    const float* inrow = in + ((size_t)(b * H_ + oh)) * (W_ * C_);

    // ---- stage im2col tile, fragment order; K and (kh,kw,c) compile-time ----
    {
        const int ow = tid & 127;
        const float* p = inrow + ow * C_;
        const bool ok = (ow < WO_);     // pad pixels (124..127) are zero
        #pragma unroll
        for (int blk = 0; blk < 12; ++blk) {
            v8h pack;
            #pragma unroll
            for (int e = 0; e < 8; ++e) {
                const int K = blk * 8 + e;
                float v = 0.f;
                if (K < KDIM && ok) {
                    const int kh = K / 15, r = K % 15;
                    const int kw = r / C_, c = r % C_;
                    v = p[kh * (W_ * C_) + kw * C_ + c];   // constant offset
                }
                pack[e] = (_Float16)v;
            }
            *(v8h*)(s_a + ow * KPAD + blkslot(blk) * 8) = pack;
        }
    }

    // ---- stage weights, fragment order, zero-padded K ----
    {
        const int n    = tid & 63;
        const int bhal = tid >> 6;            // 2 thread-halves x 6 blocks
        #pragma unroll
        for (int j = 0; j < 6; ++j) {
            const int blk = bhal * 6 + j;
            v8h pack;
            #pragma unroll
            for (int e = 0; e < 8; ++e) {
                const int K = blk * 8 + e;
                pack[e] = (K < KDIM) ? (_Float16)w_mu[K * KN_ + n]
                                     : (_Float16)0.f;
            }
            *(v8h*)(s_wf + n * KPAD + blkslot(blk) * 8) = pack;
        }
    }

    // ---- per-(row,x) channel square sums (b96 loads, no atomics) ----
    for (int i = tid; i < KS_ * W_; i += 128) {   // i = kh*128 + x
        const float* p = inrow + i * C_;
        const float v0 = p[0], v1 = p[1], v2 = p[2];
        s_sq[i] = v0 * v0 + v1 * v1 + v2 * v2;
    }
    if (tid < KN_) s_sp[tid] = softplus_f(w_sigma[tid]);
    __syncthreads();

    if (tid < WO_) {
        float s = 0.f;
        #pragma unroll
        for (int kh = 0; kh < KS_; ++kh)
            #pragma unroll
            for (int kw = 0; kw < KS_; ++kw)
                s += s_sq[kh * W_ + tid + kw];
        s_ss[tid] = s;
    }
    __syncthreads();

    // ------------------- implicit GEMM: M=128(124), N=64, K=96 -------------
    const int wave = tid >> 5;
    const int lane = tid & 31;
    const int lo   = lane & 15;
    const int hi   = lane >> 4;

    v8f acc[2][4];
    const v8f zero8 = {0.f, 0.f, 0.f, 0.f, 0.f, 0.f, 0.f, 0.f};
    #pragma unroll
    for (int mt = 0; mt < 2; ++mt)
        #pragma unroll
        for (int nt = 0; nt < 4; ++nt) acc[mt][nt] = zero8;

    #pragma unroll
    for (int kc = 0; kc < 3; ++kc) {
        const int cbase = (kc * 4 + hi * 2) * 8;    // f16 offset of half-frags
        // B fragments: two ds_load_b128 each
        v16h bf[4];
        #pragma unroll
        for (int nt = 0; nt < 4; ++nt) {
            const _Float16* wb = s_wf + (nt * 16 + lo) * KPAD + cbase;
            const v8h b0 = *(const v8h*)(wb);
            const v8h b1 = *(const v8h*)(wb + 8);
            bf[nt] = __builtin_shufflevector(b0, b1, 0, 1, 2, 3, 4, 5, 6, 7,
                                             8, 9, 10, 11, 12, 13, 14, 15);
        }
        #pragma unroll
        for (int mt = 0; mt < 2; ++mt) {
            const int mtile = wave * 2 + mt;
            const _Float16* ab = s_a + (mtile * 16 + lo) * KPAD + cbase;
            const v8h a0 = *(const v8h*)(ab);
            const v8h a1 = *(const v8h*)(ab + 8);
            const v16h af = __builtin_shufflevector(a0, a1, 0, 1, 2, 3, 4, 5,
                                                    6, 7, 8, 9, 10, 11, 12,
                                                    13, 14, 15);
            #pragma unroll
            for (int nt = 0; nt < 4; ++nt)
                acc[mt][nt] = __builtin_amdgcn_wmma_f32_16x16x32_f16(
                    false, af, false, bf[nt], (short)0, acc[mt][nt],
                    false, false);
        }
    }

    // ---- store mu_out (C-layout: M = r + 8*hi, N = lo) ----
    const size_t pixbase = ((size_t)(b * HO_ + oh)) * WO_ * KN_;
    #pragma unroll
    for (int mt = 0; mt < 2; ++mt) {
        const int mtile = wave * 2 + mt;
        #pragma unroll
        for (int nt = 0; nt < 4; ++nt) {
            const int n = nt * 16 + lo;
            #pragma unroll
            for (int r = 0; r < 8; ++r) {
                const int ow = mtile * 16 + r + hi * 8;
                if (ow < WO_)
                    mu_out[pixbase + (size_t)ow * KN_ + n] = acc[mt][nt][r];
            }
        }
    }

    // ---- Sigma_out: coalesced flat stream ----
    const float inv_xdim = 1.f / (float)(KS_ * KS_ * C_);
    for (int i = tid; i < WO_ * KN_; i += 128) {
        const int ow = i >> 6;
        const int n  = i & 63;
        sigma_out[pixbase + i] = s_sp[n] * s_ss[ow] * inv_xdim;
    }
}

// ---------------------------------------------------------------------------
extern "C" void kernel_launch(void* const* d_in, const int* in_sizes, int n_in,
                              void* d_out, int out_size, void* d_ws, size_t ws_size,
                              hipStream_t stream) {
    const float* mu_in = (const float*)d_in[0];   // 128*128*128*3
    const float* w_mu  = (const float*)d_in[1];   // 5*5*3*64
    const float* w_sig = (const float*)d_in[2];   // 64
    float* out = (float*)d_out;

    const size_t TOT = (size_t)NB * HO_ * WO_ * KN_;   // 125,960,192
    float* mu_out = out;
    float* sg_out = out + TOT;
    float* kl_out = out + 2 * TOT;

    kl_kernel<<<1, 64, 0, stream>>>(w_mu, w_sig, kl_out);

    dim3 grid(HO_, NB);   // 124 x 128 blocks
    vdp_conv_kernel<<<grid, 128, 0, stream>>>(mu_in, w_mu, w_sig, mu_out, sg_out);
}
